// SpatialGcnMotion_11708080849106
// MI455X (gfx1250) — matrix operand order repaired and checked by reading, compiled, not run on hardware
//
#include <hip/hip_runtime.h>
#include <hip/hip_bf16.h>
#include <stdint.h>

// ---------------- problem constants ----------------
#define N_    64
#define CIN_  64
#define T_    256
#define V_    25
#define COUT_ 128
#define CI_   64
constexpr int TV   = T_ * V_;       // 6400 columns per (n)
constexpr float EPSB = 1e-5f;

// ---------------- vector types ----------------
typedef __attribute__((ext_vector_type(16))) __bf16   v16bf;
typedef __attribute__((ext_vector_type(8)))  __bf16   v8bf;
typedef __attribute__((ext_vector_type(8)))  float    v8f;
typedef __attribute__((ext_vector_type(4)))  uint32_t u32x4;
typedef __attribute__((ext_vector_type(8)))  int      i32x8;
typedef __attribute__((ext_vector_type(4)))  int      i32x4;

// ---------------- workspace layout (bytes) ----------------
// stats: [0..127]=sumY [128..255]=sqY [256..383]=sumD [384..511]=sqD
//        [512..639]=scaleY [640..767]=biasY [768..895]=scaleD [896..1023]=biasD
constexpr size_t ST_OFF   = 0;
constexpr size_t XH_OFF   = 4096;                                  // bf16 x  [n][t*v][i]
constexpr size_t XH_SZ    = (size_t)N_ * TV * CIN_ * 2;            // 52,428,800
constexpr size_t Y_OFF    = XH_OFF + XH_SZ;                        // f32 y/y2 [n][c][t][v]
constexpr size_t Y_SZ     = (size_t)N_ * COUT_ * TV * 4;           // 209,715,200
constexpr size_t D_OFF    = Y_OFF + Y_SZ;                          // f32 d    [n][c][t][v]
constexpr size_t RD_OFF   = D_OFF + Y_SZ;                          // f32 Rd   [n][c][u][w]
constexpr size_t RD_SZ    = (size_t)N_ * COUT_ * V_ * V_ * 4;      // 20,480,000
constexpr size_t DH_OFF   = RD_OFF + RD_SZ;                        // bf16 diff [n][v][w][i]
constexpr size_t DH_SZ    = (size_t)N_ * V_ * V_ * CIN_ * 2;       // 5,120,000
constexpr size_t XMF_OFF  = DH_OFF + DH_SZ;                        // f32 [n][i][v]
constexpr size_t XMB_OFF  = XMF_OFF + 409600;
constexpr size_t XF_OFF   = XMB_OFF + 409600;                      // f32 [n][o][v]
constexpr size_t XB_OFF   = XF_OFF + 409600;
constexpr size_t G_OFF    = XB_OFF + 409600;                       // f32 A+Al (1875)
constexpr size_t WPH_OFF  = G_OFF + 7680;                          // bf16 Wp   (24576 el)
constexpr size_t WM3H_OFF = WPH_OFF + 49152;                       // bf16 Wm3  (8192 el)
constexpr size_t WM4H_OFF = WM3H_OFF + 16384;                      // bf16 Wm4
constexpr size_t DWH_OFF  = WM4H_OFF + 16384;                      // bf16 down_W

// ---------------- helpers ----------------
__device__ __forceinline__ v8f vzero8() {
  v8f z;
#pragma unroll
  for (int r = 0; r < 8; ++r) z[r] = 0.f;
  return z;
}

// A-matrix lane fragment for 16x16x32 bf16: two 8-element
// (16B) chunks per lane (callers pass k = half*16..+7 and half*16+8..+15: contiguous, same order as B).
__device__ __forceinline__ v16bf ld_a_frag(const __bf16* p0, const __bf16* p1) {
  v8bf a = *(const v8bf*)p0;
  v8bf b = *(const v8bf*)p1;
  v16bf r;
#pragma unroll
  for (int j = 0; j < 8; ++j) { r[j] = a[j]; r[j + 8] = b[j]; }
  return r;
}

__device__ __forceinline__ v8f wmma_bf16(v16bf a, v16bf b, v8f c) {
  return __builtin_amdgcn_wmma_f32_16x16x32_bf16(false, a, false, b, (short)0, c,
                                                 false, false);
}

// Tensor Data Mover: 1D contiguous bf16-element copy, global -> LDS.
// Wave-level instruction; caller gates to one wave and waits TENSORcnt.
__device__ __forceinline__ void tdm_load_lds(void* lds_dst, const void* gsrc,
                                             unsigned nbytes) {
  unsigned elems = nbytes >> 1;  // 2-byte elements
  unsigned long long ga = (unsigned long long)(uintptr_t)gsrc;
  unsigned ldsoff = (unsigned)(uintptr_t)lds_dst;  // flat->LDS truncates to [31:0]
  u32x4 g0;
  g0[0] = 1u;                                            // count=1, user D#
  g0[1] = ldsoff;                                        // lds_addr
  g0[2] = (unsigned)(ga & 0xFFFFFFFFu);                  // global_addr lo
  g0[3] = (unsigned)((ga >> 32) & 0x1FFFFFFu) | (2u << 30);  // addr hi | type=2
  i32x8 g1;
  g1[0] = (int)(1u << 16);                               // data_size = 2B
  g1[1] = (int)((elems & 0xFFFFu) << 16);                // tensor_dim0 lo16
  g1[2] = (int)(((elems >> 16) & 0xFFFFu) | (1u << 16)); // dim0 hi16 | tensor_dim1=1 lo
  g1[3] = (int)((elems & 0xFFFFu) << 16);                // dim1 hi=0 | tile_dim0
  g1[4] = 1;                                             // tile_dim1=1, tile_dim2=0
  g1[5] = (int)elems;                                    // tensor_dim0_stride lo32
  g1[6] = 0;
  g1[7] = 0;
  i32x4 gz4; gz4[0] = gz4[1] = gz4[2] = gz4[3] = 0;
  i32x8 gz8;
#pragma unroll
  for (int j = 0; j < 8; ++j) gz8[j] = 0;
  __builtin_amdgcn_tensor_load_to_lds(g0, g1, gz4, gz4, gz8, 0);
}

// ================= k0: zero stats, bf16 weights, G = A+Al =================
__global__ void k0_misc(char* ws, const float* Wp, const float* Wm3,
                        const float* Wm4, const float* dW, const float* A,
                        const float* Al) {
  float* gst  = (float*)(ws + ST_OFF);
  __bf16* wph = (__bf16*)(ws + WPH_OFF);
  __bf16* w3h = (__bf16*)(ws + WM3H_OFF);
  __bf16* w4h = (__bf16*)(ws + WM4H_OFF);
  __bf16* dwh = (__bf16*)(ws + DWH_OFF);
  float* G    = (float*)(ws + G_OFF);
  int tid = blockIdx.x * blockDim.x + threadIdx.x;
  int stride = gridDim.x * blockDim.x;
  for (int i = tid; i < 1024; i += stride) gst[i] = 0.f;
  for (int i = tid; i < 3 * COUT_ * CIN_; i += stride) wph[i] = (__bf16)Wp[i];
  for (int i = tid; i < COUT_ * CI_; i += stride) w3h[i] = (__bf16)Wm3[i];
  for (int i = tid; i < COUT_ * CIN_; i += stride) w4h[i] = (__bf16)Wm4[i];
  for (int i = tid; i < COUT_ * CIN_; i += stride) dwh[i] = (__bf16)dW[i];
  for (int i = tid; i < 3 * V_ * V_; i += stride) G[i] = A[i] + Al[i];
}

// ================= k1: transpose x (n,i,tv) -> bf16 (n,tv,i) =================
__global__ __launch_bounds__(256) void k1_prep(const float* x, char* ws) {
  __shared__ float tile[64][65];
  __bf16* xh = (__bf16*)(ws + XH_OFF);
  int n = blockIdx.x, jb = blockIdx.y * 64;
  int tid = threadIdx.x;
  for (int idx = tid; idx < 4096; idx += 256) {
    int i = idx >> 6, j = idx & 63;
    tile[j][i] = x[((size_t)n * CIN_ + i) * TV + jb + j];
  }
  __syncthreads();
  for (int idx = tid; idx < 4096; idx += 256) {
    int j = idx >> 6, i = idx & 63;
    xh[((size_t)n * TV + jb + j) * CIN_ + i] = (__bf16)tile[j][i];
  }
}

// ================= k2: temporal means (mean commutes with 1x1 conv) ========
__global__ void k2_means(const float* x, char* ws) {
  float* xmf = (float*)(ws + XMF_OFF);
  float* xmb = (float*)(ws + XMB_OFF);
  int idx = blockIdx.x * blockDim.x + threadIdx.x;
  if (idx >= N_ * CIN_ * V_) return;
  int n = idx / (CIN_ * V_), rem = idx % (CIN_ * V_);
  int i = rem / V_, v = rem % V_;
  const float* p = x + ((size_t)n * CIN_ + i) * TV + v;
  float sf = 0.f, sb = 0.f;
  for (int t = 0; t < T_ - 2; ++t) sf += p[t * V_];
  for (int t = 2; t < T_; ++t) sb += p[t * V_];
  xmf[idx] = sf * (1.f / (T_ - 2));
  xmb[idx] = sb * (1.f / (T_ - 2));
}

// ================= k3a: xf = Wm1*xmf + bm1 ; xb = Wm2*xmb + bm2 ============
__global__ void k3a_small(char* ws, const float* Wm1, const float* bm1,
                          const float* Wm2, const float* bm2) {
  const float* xmf = (const float*)(ws + XMF_OFF);
  const float* xmb = (const float*)(ws + XMB_OFF);
  float* xf = (float*)(ws + XF_OFF);
  float* xb = (float*)(ws + XB_OFF);
  int idx = blockIdx.x * blockDim.x + threadIdx.x;
  if (idx >= N_ * CI_ * V_) return;
  int n = idx / (CI_ * V_), rem = idx % (CI_ * V_);
  int o = rem / V_, v = rem % V_;
  float a = bm1[o], b = bm2[o];
  for (int i = 0; i < CIN_; ++i) {
    a += Wm1[o * CIN_ + i] * xmf[((size_t)n * CIN_ + i) * V_ + v];
    b += Wm2[o * CIN_ + i] * xmb[((size_t)n * CIN_ + i) * V_ + v];
  }
  xf[idx] = a;
  xb[idx] = b;
}

// ================= k3b: diffh[n][v][w][i] = bf16(xb[n,i,v] - xf[n,i,w]) ====
__global__ void k3b_diff(char* ws) {
  const float* xf = (const float*)(ws + XF_OFF);
  const float* xb = (const float*)(ws + XB_OFF);
  __bf16* dh = (__bf16*)(ws + DH_OFF);
  int idx = blockIdx.x * blockDim.x + threadIdx.x;
  if (idx >= N_ * V_ * V_) return;
  int n = idx / (V_ * V_), rem = idx % (V_ * V_);
  int v = rem / V_, w = rem % V_;
  __bf16* d = dh + (size_t)idx * CI_;
  for (int i = 0; i < CI_; ++i)
    d[i] = (__bf16)(xb[((size_t)n * CI_ + i) * V_ + v] -
                    xf[((size_t)n * CI_ + i) * V_ + w]);
}

// ================= k4: Rd = tanh(Wm3 x diff + bm3)  (WMMA) =================
__global__ __launch_bounds__(256) void k4_rd(char* ws, const float* bm3) {
  const __bf16* w3h = (const __bf16*)(ws + WM3H_OFF);
  const __bf16* dh  = (const __bf16*)(ws + DH_OFF);
  float* Rd = (float*)(ws + RD_OFF);
  int wave = threadIdx.x >> 5, lane = threadIdx.x & 31;
  int half = lane >> 4, ln = lane & 15;
  int ctile = wave;                              // 8 waves -> 8 c-tiles
  int col = blockIdx.x * 16 + ln;                // 2500 col tiles -> 40000 cols
  int m = ctile * 16 + ln;
  v8f acc = vzero8();
  const __bf16* wr = w3h + (size_t)m * CI_;
#pragma unroll
  for (int kb = 0; kb < CI_; kb += 32) {
    v16bf a = ld_a_frag(wr + kb + half * 16, wr + kb + half * 16 + 8);
    v16bf b = *(const v16bf*)(dh + (size_t)col * CI_ + kb + half * 16);
    acc = wmma_bf16(a, b, acc);
  }
  int n = col / (V_ * V_), uw = col % (V_ * V_);
#pragma unroll
  for (int r = 0; r < 8; ++r) {
    int mm = ctile * 16 + r + 8 * half;
    Rd[((size_t)n * COUT_ + mm) * (V_ * V_) + uw] = tanhf(acc[r] + bm3[mm]);
  }
}

// ====== k5: y = Σ_s Wp_s x (G_s ⊗ x) (+bias corr), d = down_W x x  (WMMA) ==
// grid (tb=16, n=64); dyn LDS: xt bf16 51200 | xg bf16 153600 | gsm 7500 |
// gsum 300 | bps 1536 | sD/sD2 1024  => 215232 B
__global__ __launch_bounds__(256) void k5_graphconv(char* ws, const float* bp) {
  extern __shared__ char sm[];
  __bf16* xt  = (__bf16*)sm;
  __bf16* xg  = (__bf16*)(sm + 51200);
  float* gsm  = (float*)(sm + 204800);
  float* gsum = (float*)(sm + 212352);
  float* bps  = (float*)(sm + 212672);
  float* sD   = (float*)(sm + 214208);
  float* sD2  = (float*)(sm + 214720);

  int tb = blockIdx.x, n = blockIdx.y;
  int tid = threadIdx.x, wave = tid >> 5, lane = tid & 31;
  int half = lane >> 4, ln = lane & 15;

  const __bf16* xh  = (const __bf16*)(ws + XH_OFF);
  const __bf16* wph = (const __bf16*)(ws + WPH_OFF);
  const __bf16* dwh = (const __bf16*)(ws + DWH_OFF);
  const float* G    = (const float*)(ws + G_OFF);
  float* yw  = (float*)(ws + Y_OFF);
  float* dw  = (float*)(ws + D_OFF);
  float* gst = (float*)(ws + ST_OFF);

  for (int i = tid; i < 3 * V_ * V_; i += 256) gsm[i] = G[i];
  for (int i = tid; i < 3 * COUT_; i += 256) bps[i] = bp[i];
  for (int i = tid; i < COUT_; i += 256) { sD[i] = 0.f; sD2[i] = 0.f; }
  if (wave == 0) {  // one wave issues the TDM for the 400x64 bf16 x-tile
    tdm_load_lds(xt, xh + ((size_t)n * TV + tb * 400) * CIN_, 400 * CIN_ * 2);
    __builtin_amdgcn_s_wait_tensorcnt(0);
  }
  __syncthreads();

  // row sums of G_s (bias correction for bp through the graph contraction)
  for (int i = tid; i < 3 * V_; i += 256) {
    int s = i / V_, u = i % V_;
    float a = 0.f;
    for (int v = 0; v < V_; ++v) a += gsm[s * (V_ * V_) + u * V_ + v];
    gsum[i] = a;
  }
  // xg_s[tl,u,i] = Σ_v G_s[u,v] * x[tl,v,i]   (graph contraction on K side)
  for (int idx = tid; idx < 3 * 400 * CIN_; idx += 256) {
    int s = idx / (400 * CIN_), rem = idx % (400 * CIN_);
    int tl = rem / (V_ * CIN_), r2 = rem % (V_ * CIN_);
    int u = r2 / CIN_, i = r2 % CIN_;
    const float* gr = gsm + s * (V_ * V_) + u * V_;
    const __bf16* xr = xt + (size_t)(tl * V_) * CIN_ + i;
    float acc = 0.f;
#pragma unroll 5
    for (int v = 0; v < V_; ++v) acc += gr[v] * (float)xr[v * CIN_];
    xg[((size_t)(s * 400 + tl * V_ + u)) * CIN_ + i] = (__bf16)acc;
  }
  __syncthreads();

  for (int tile = wave; tile < 200; tile += 8) {
    int ctile = tile / 25, colt = tile % 25;
    int m = ctile * 16 + ln;
    int col = colt * 16 + ln;
    v8f ay = vzero8(), ad = vzero8();
#pragma unroll
    for (int s = 0; s < 3; ++s) {
      const __bf16* wr = wph + ((size_t)(s * COUT_ + m)) * CIN_;
#pragma unroll
      for (int kb = 0; kb < CIN_; kb += 32) {
        v16bf a = ld_a_frag(wr + kb + half * 16, wr + kb + half * 16 + 8);
        v16bf b = *(const v16bf*)(xg + ((size_t)(s * 400 + col)) * CIN_ + kb +
                                  half * 16);
        ay = wmma_bf16(a, b, ay);
      }
    }
    {
      const __bf16* wr = dwh + (size_t)m * CIN_;
#pragma unroll
      for (int kb = 0; kb < CIN_; kb += 32) {
        v16bf a = ld_a_frag(wr + kb + half * 16, wr + kb + half * 16 + 8);
        v16bf b = *(const v16bf*)(xt + (size_t)col * CIN_ + kb + half * 16);
        ad = wmma_bf16(a, b, ad);
      }
    }
    int tl = col / V_, u = col % V_;
    int tg = tb * 16 + tl;
#pragma unroll
    for (int r = 0; r < 8; ++r) {
      int mm = ctile * 16 + r + 8 * half;
      float bc = bps[mm] * gsum[u] + bps[COUT_ + mm] * gsum[V_ + u] +
                 bps[2 * COUT_ + mm] * gsum[2 * V_ + u];
      size_t oi = ((size_t)(n * COUT_ + mm) * T_ + tg) * V_ + u;
      yw[oi] = ay[r] + bc;
      float dv = ad[r];
      dw[oi] = dv;
      atomicAdd(&sD[mm], dv);
      atomicAdd(&sD2[mm], dv * dv);
    }
  }
  __syncthreads();
  for (int c = tid; c < COUT_; c += 256) {
    atomicAdd(&gst[256 + c], sD[c]);
    atomicAdd(&gst[384 + c], sD2[c]);
  }
}

// ====== k6: Xm = Wm4 x x + bm4 (WMMA, LDS) ; z = Rd·Xm ; y2 = αz + y =======
// grid (tb=16, n=64); dyn LDS: xt 51200 | Xm f32 204800 | sY/sY2 1024 => 257024
__global__ __launch_bounds__(256) void k6_dyn(char* ws, const float* bm4,
                                              const float* alpha) {
  extern __shared__ char sm[];
  __bf16* xt = (__bf16*)sm;
  float* Xm  = (float*)(sm + 51200);
  float* sY  = (float*)(sm + 256000);
  float* sY2 = (float*)(sm + 256512);

  int tb = blockIdx.x, n = blockIdx.y;
  int tid = threadIdx.x, wave = tid >> 5, lane = tid & 31;
  int half = lane >> 4, ln = lane & 15;

  const __bf16* xh  = (const __bf16*)(ws + XH_OFF);
  const __bf16* w4h = (const __bf16*)(ws + WM4H_OFF);
  const float* Rd   = (const float*)(ws + RD_OFF);
  float* yw  = (float*)(ws + Y_OFF);
  float* gst = (float*)(ws + ST_OFF);

  for (int i = tid; i < COUT_; i += 256) { sY[i] = 0.f; sY2[i] = 0.f; }
  if (wave == 0) {
    tdm_load_lds(xt, xh + ((size_t)n * TV + tb * 400) * CIN_, 400 * CIN_ * 2);
    __builtin_amdgcn_s_wait_tensorcnt(0);
  }
  __syncthreads();

  for (int tile = wave; tile < 200; tile += 8) {
    int ctile = tile / 25, colt = tile % 25;
    int m = ctile * 16 + ln;
    int col = colt * 16 + ln;
    v8f acc = vzero8();
    const __bf16* wr = w4h + (size_t)m * CIN_;
#pragma unroll
    for (int kb = 0; kb < CIN_; kb += 32) {
      v16bf a = ld_a_frag(wr + kb + half * 16, wr + kb + half * 16 + 8);
      v16bf b = *(const v16bf*)(xt + (size_t)col * CIN_ + kb + half * 16);
      acc = wmma_bf16(a, b, acc);
    }
#pragma unroll
    for (int r = 0; r < 8; ++r) {
      int mm = ctile * 16 + r + 8 * half;
      Xm[mm * 400 + col] = acc[r] + bm4[mm];
    }
  }
  __syncthreads();

  float alph = alpha[0];
  for (int idx = tid; idx < COUT_ * 400; idx += 256) {
    int c = idx / 400, rem = idx % 400;
    int tl = rem / V_, u = rem % V_;
    const float* rr = Rd + ((size_t)(n * COUT_ + c)) * (V_ * V_) + u * V_;
    const float* xm = Xm + c * 400 + tl * V_;
    float z = 0.f;
#pragma unroll 5
    for (int v = 0; v < V_; ++v) z += rr[v] * xm[v];
    size_t oi = ((size_t)(n * COUT_ + c) * T_ + tb * 16 + tl) * V_ + u;
    float y2 = alph * z + yw[oi];
    yw[oi] = y2;
    atomicAdd(&sY[c], y2);
    atomicAdd(&sY2[c], y2 * y2);
  }
  __syncthreads();
  for (int c = tid; c < COUT_; c += 256) {
    atomicAdd(&gst[c], sY[c]);
    atomicAdd(&gst[128 + c], sY2[c]);
  }
}

// ================= k7: finalize BN scale/bias per channel =================
__global__ void k7_stats(char* ws, const float* g1, const float* b1,
                         const float* g2, const float* b2) {
  float* gst = (float*)(ws + ST_OFF);
  int c = threadIdx.x;
  if (c >= COUT_) return;
  const float cnt = (float)((size_t)N_ * TV);
  float mY = gst[c] / cnt;
  float vY = gst[128 + c] / cnt - mY * mY;
  float iY = rsqrtf(vY + EPSB);
  float scY = g1[c] * iY;
  gst[512 + c] = scY;
  gst[640 + c] = b1[c] - mY * scY;
  float mD = gst[256 + c] / cnt;
  float vD = gst[384 + c] / cnt - mD * mD;
  float iD = rsqrtf(vD + EPSB);
  float scD = g2[c] * iD;
  gst[768 + c] = scD;
  gst[896 + c] = b2[c] - mD * scD;
}

// ================= k8: out = relu(bn(y2) + bn(d))  (float4) =================
__global__ __launch_bounds__(256) void k8_out(const char* ws, float* out) {
  const float* gst = (const float*)(ws + ST_OFF);
  const float4* yv4 = (const float4*)(ws + Y_OFF);
  const float4* dv4 = (const float4*)(ws + D_OFF);
  size_t idx = (size_t)blockIdx.x * 256 + threadIdx.x;   // 13,107,200 vec4s
  size_t e = idx * 4;
  int c = (int)((e / TV) % COUT_);
  float scY = gst[512 + c], biY = gst[640 + c];
  float scD = gst[768 + c], biD = gst[896 + c];
  float4 y = yv4[idx], d = dv4[idx];
  float4 o;
  o.x = fmaxf(y.x * scY + biY + d.x * scD + biD, 0.f);
  o.y = fmaxf(y.y * scY + biY + d.y * scD + biD, 0.f);
  o.z = fmaxf(y.z * scY + biY + d.z * scD + biD, 0.f);
  o.w = fmaxf(y.w * scY + biY + d.w * scD + biD, 0.f);
  ((float4*)out)[idx] = o;
}

// ================= host side =================
extern "C" void kernel_launch(void* const* d_in, const int* in_sizes, int n_in,
                              void* d_out, int out_size, void* d_ws,
                              size_t ws_size, hipStream_t stream) {
  const float* x     = (const float*)d_in[0];
  const float* A     = (const float*)d_in[1];
  const float* Al    = (const float*)d_in[2];
  const float* Wp    = (const float*)d_in[3];
  const float* bp    = (const float*)d_in[4];
  const float* Wm1   = (const float*)d_in[5];
  const float* bm1   = (const float*)d_in[6];
  const float* Wm2   = (const float*)d_in[7];
  const float* bm2   = (const float*)d_in[8];
  const float* Wm3   = (const float*)d_in[9];
  const float* bm3   = (const float*)d_in[10];
  const float* Wm4   = (const float*)d_in[11];
  const float* bm4   = (const float*)d_in[12];
  const float* alpha = (const float*)d_in[13];
  const float* bng   = (const float*)d_in[14];
  const float* bnb   = (const float*)d_in[15];
  const float* dWg   = (const float*)d_in[16];
  const float* dg    = (const float*)d_in[17];
  const float* db    = (const float*)d_in[18];
  float* out = (float*)d_out;
  char* ws = (char*)d_ws;

  k0_misc<<<64, 256, 0, stream>>>(ws, Wp, Wm3, Wm4, dWg, A, Al);
  k1_prep<<<dim3(N_, TV / 64), 256, 0, stream>>>(x, ws);
  k2_means<<<(N_ * CIN_ * V_ + 255) / 256, 256, 0, stream>>>(x, ws);
  k3a_small<<<(N_ * CI_ * V_ + 255) / 256, 256, 0, stream>>>(ws, Wm1, bm1, Wm2,
                                                             bm2);
  k3b_diff<<<(N_ * V_ * V_ + 255) / 256, 256, 0, stream>>>(ws);
  k4_rd<<<(N_ * V_ * V_) / 16, 256, 0, stream>>>(ws, bm3);          // 2500 blocks
  k5_graphconv<<<dim3(16, N_), 256, 215232, stream>>>(ws, bp);
  k6_dyn<<<dim3(16, N_), 256, 257024, stream>>>(ws, bm4, alpha);
  k7_stats<<<1, 128, 0, stream>>>(ws, bng, bnb, dg, db);
  k8_out<<<(unsigned)((size_t)N_ * COUT_ * TV / 4 / 256), 256, 0, stream>>>(ws,
                                                                            out);
}
